// ConvMultiheadAttention_22539988369988
// MI455X (gfx1250) — compile-verified, matrix operand order checked by reading, not run on hardware
//
#include <hip/hip_runtime.h>
#include <stdint.h>

// ---------------- problem constants ----------------
#define D_MODEL   512
#define NUM_HEADS 8
#define HEAD_DIM  64
#define SEQ       1024
#define SEQP      (SEQ + 2)     // zero-padded rows: [0] and [SEQ+1] are zeros
#define BATCH     8

typedef __attribute__((ext_vector_type(16))) _Float16 v16h;
typedef __attribute__((ext_vector_type(8)))  float    v8f;

union Frag16 { v16h h; uint32_t u[8]; };

// Load a 16x32 f16 WMMA fragment (A: MxK rows, or B: KxN cols -- same lane map).
// ISA layout: lane&15 = row/col index; lane>>4 selects K-half; VGPR v holds the
// packed f16 pair k = (v<4 ? 0 : 16) + 8*(lane>>4) + 2*(v&3).
// `tile` points at (row 0, k 0); K-dim must be contiguous in memory.
// The 4+4 contiguous u32s vectorize into two global_load_b128.
__device__ inline v16h ld_frag(const _Float16* tile, int stride) {
  const int lane = threadIdx.x & 31;
  const uint32_t* row = (const uint32_t*)(tile + (size_t)(lane & 15) * stride);
  const int kb = (lane >> 4) * 4;   // u32 units
  Frag16 f;
#pragma unroll
  for (int v = 0; v < 4; ++v) {
    f.u[v]     = row[kb + v];
    f.u[v + 4] = row[kb + 8 + v];
  }
  return f.h;
}

// ---------------- format conversion kernels ----------------
// f32 [B][S][D] -> f16 padded [B][S+2][D]; rows 0 and S+1 of each batch are zero
// so 3-tap conv A-fragments never need bounds checks.
__global__ __launch_bounds__(256) void cvt_pad_kernel(const float* __restrict__ s,
                                                      _Float16* __restrict__ d, int n) {
  int i = blockIdx.x * 256 + threadIdx.x;
  if (i >= n) return;
  int c = i & (D_MODEL - 1);
  int r = (i / D_MODEL) % SEQP;
  int b = i / (D_MODEL * SEQP);
  float v = 0.0f;
  if (r != 0 && r != SEQ + 1)
    v = s[((size_t)b * SEQ + (r - 1)) * D_MODEL + c];
  d[i] = (_Float16)v;
}

__global__ __launch_bounds__(256) void cvt_f16_kernel(const float* __restrict__ s,
                                                      _Float16* __restrict__ d, int n) {
  int i = blockIdx.x * 256 + threadIdx.x;
  if (i < n) d[i] = (_Float16)s[i];
}

// Wq/Wk: [co][ci][3] f32  ->  [tap][co][ci] f16 (K-dim = ci contiguous for B-frags)
__global__ __launch_bounds__(256) void repack_w3_kernel(const float* __restrict__ s,
                                                        _Float16* __restrict__ d) {
  int i = blockIdx.x * 256 + threadIdx.x;
  if (i >= D_MODEL * D_MODEL) return;
#pragma unroll
  for (int kk = 0; kk < 3; ++kk)
    d[kk * D_MODEL * D_MODEL + i] = (_Float16)s[(size_t)i * 3 + kk];
}

// ---------------- fused conv-projection / GEMM kernel ----------------
// X is the PADDED activation buffer [B][S+2][D]; data row s lives at r = s+1.
// out[m=(b,s)][co] = bias[co] + sum_tap sum_ci X[b, s+tap-taps/2, ci] * W[tap][co][ci]
// One wave computes a 32(M) x 64(N) tile: two M sub-tiles share every B fragment,
// doubling WMMAs per weight load.
// mode 0: f16 out [b,h,s,d]        (Q, K)
// mode 1: f16 out [b,h,d,s]        (V, transposed for B-fragments of P@V)
// mode 2: f32 out flat [b,s,D]     (final out-projection)
__global__ __launch_bounds__(32) void proj_kernel(const _Float16* __restrict__ X,
                                                  const _Float16* __restrict__ W,
                                                  const float* __restrict__ bias,
                                                  void* __restrict__ out,
                                                  int taps, int mode) {
  const int mt   = blockIdx.x;            // (b, 32-row s-tile): 32 tiles per batch
  const int n0   = blockIdx.y * 64;
  const int b    = mt >> 5;
  const int s0   = (mt & 31) << 5;
  const int lane = threadIdx.x & 31;
  const int col  = lane & 15, half = lane >> 4;

  v8f c[2][4];
#pragma unroll
  for (int nt = 0; nt < 4; ++nt) {
    float bv = bias[n0 + nt * 16 + col];
#pragma unroll
    for (int i = 0; i < 8; ++i) { c[0][nt][i] = bv; c[1][nt][i] = bv; }
  }

  const _Float16* xb = X + (size_t)b * SEQP * D_MODEL;
  for (int ci = 0; ci < D_MODEL; ci += 32) {
    for (int kk = 0; kk < taps; ++kk) {
      const int r0 = s0 + kk - (taps >> 1) + 1;   // padded row of sub-tile 0
      v16h a0 = ld_frag(xb + (size_t)r0 * D_MODEL + ci, D_MODEL);
      v16h a1 = ld_frag(xb + (size_t)(r0 + 16) * D_MODEL + ci, D_MODEL);
      const _Float16* wb = W + (size_t)kk * D_MODEL * D_MODEL + ci;
#pragma unroll
      for (int nt = 0; nt < 4; ++nt) {
        v16h bf = ld_frag(wb + (size_t)(n0 + nt * 16) * D_MODEL, D_MODEL);
        c[0][nt] = __builtin_amdgcn_wmma_f32_16x16x32_f16(false, a0, false, bf,
                                                          (short)0, c[0][nt], false, false);
        c[1][nt] = __builtin_amdgcn_wmma_f32_16x16x32_f16(false, a1, false, bf,
                                                          (short)0, c[1][nt], false, false);
      }
    }
  }

#pragma unroll
  for (int mi = 0; mi < 2; ++mi) {
#pragma unroll
    for (int nt = 0; nt < 4; ++nt) {
#pragma unroll
      for (int i = 0; i < 8; ++i) {
        const int s  = s0 + mi * 16 + i + 8 * half;  // C layout: VGPR i -> M = i + 8*half
        const int co = n0 + nt * 16 + col;
        const float v = c[mi][nt][i];
        if (mode == 2) {
          ((float*)out)[((size_t)b * SEQ + s) * D_MODEL + co] = v;
        } else {
          const int h = co >> 6, dd = co & 63;
          const size_t addr = (mode == 0)
              ? (((size_t)(b * NUM_HEADS + h) * SEQ + s) * HEAD_DIM + dd)
              : (((size_t)(b * NUM_HEADS + h) * HEAD_DIM + dd) * SEQ + s);
          ((_Float16*)out)[addr] = (_Float16)v;
        }
      }
    }
  }
}

// ---------------- fused flash attention + head-averaged attention map ----------------
// Workgroup = (b, 16-row s-tile); wave w = head w. 256 threads = 8 wave32.
__global__ __launch_bounds__(256) void attn_kernel(const _Float16* __restrict__ Qh,
                                                   const _Float16* __restrict__ Kh,
                                                   const _Float16* __restrict__ Vt,
                                                   _Float16* __restrict__ Oh,
                                                   float* __restrict__ ave) {
  __shared__ float    lds_ave[16][64];
  __shared__ _Float16 lds_p[NUM_HEADS][16][64];

  const int blk  = blockIdx.x;
  const int b    = blk >> 6;
  const int s0   = (blk & 63) << 4;
  const int h    = threadIdx.x >> 5;
  const int lane = threadIdx.x & 31;
  const int col  = lane & 15, half = lane >> 4;
  const float inv_hd = 1.0f / (float)HEAD_DIM;   // scores scaled by head_dim (not sqrt)

  const _Float16* Qhd = Qh + (size_t)(b * NUM_HEADS + h) * SEQ * HEAD_DIM;
  const _Float16* Khd = Kh + (size_t)(b * NUM_HEADS + h) * SEQ * HEAD_DIM;
  const _Float16* Vhd = Vt + (size_t)(b * NUM_HEADS + h) * HEAD_DIM * SEQ;

  // Q A-fragments for this s-tile (2 k-steps over hd=64); reused across all t.
  v16h qa[2];
#pragma unroll
  for (int jj = 0; jj < 2; ++jj)
    qa[jj] = ld_frag(Qhd + (size_t)s0 * HEAD_DIM + jj * 32, HEAD_DIM);

  float mrow[8], lrow[8];
#pragma unroll
  for (int i = 0; i < 8; ++i) { mrow[i] = -1e30f; lrow[i] = 0.0f; }

  // ---- phase 1: online softmax statistics over all keys ----
  for (int t0 = 0; t0 < SEQ; t0 += 64) {
    __builtin_prefetch(Khd + (size_t)(t0 + 64) * HEAD_DIM, 0, 0);
    v8f c[4];
#pragma unroll
    for (int nt = 0; nt < 4; ++nt) { v8f z = {}; c[nt] = z; }
#pragma unroll
    for (int nt = 0; nt < 4; ++nt) {
#pragma unroll
      for (int jj = 0; jj < 2; ++jj) {
        v16h kb = ld_frag(Khd + (size_t)(t0 + nt * 16) * HEAD_DIM + jj * 32, HEAD_DIM);
        c[nt] = __builtin_amdgcn_wmma_f32_16x16x32_f16(false, qa[jj], false, kb,
                                                       (short)0, c[nt], false, false);
      }
    }
#pragma unroll
    for (int i = 0; i < 8; ++i) {  // row-wise (across 16 lanes of the half) max/sum
      float x0 = c[0][i] * inv_hd, x1 = c[1][i] * inv_hd;
      float x2 = c[2][i] * inv_hd, x3 = c[3][i] * inv_hd;
      float cm = fmaxf(fmaxf(x0, x1), fmaxf(x2, x3));
#pragma unroll
      for (int msk = 1; msk < 16; msk <<= 1) cm = fmaxf(cm, __shfl_xor(cm, msk, 32));
      float mn = fmaxf(mrow[i], cm);
      float e = __expf(x0 - mn) + __expf(x1 - mn) + __expf(x2 - mn) + __expf(x3 - mn);
#pragma unroll
      for (int msk = 1; msk < 16; msk <<= 1) e += __shfl_xor(e, msk, 32);
      lrow[i] = lrow[i] * __expf(mrow[i] - mn) + e;
      mrow[i] = mn;
    }
  }
#pragma unroll
  for (int i = 0; i < 8; ++i) lrow[i] = 1.0f / lrow[i];

  // zero the shared head-average tile
#pragma unroll
  for (int j = 0; j < 4; ++j) {
    int flat = threadIdx.x + j * 256;
    lds_ave[flat >> 6][flat & 63] = 0.0f;
  }
  __syncthreads();

  v8f oacc[4];
#pragma unroll
  for (int nt = 0; nt < 4; ++nt) { v8f z = {}; oacc[nt] = z; }

  // ---- phase 2: probabilities, P@V, head-averaged attention output ----
  for (int t0 = 0; t0 < SEQ; t0 += 64) {
    v8f c[4];
#pragma unroll
    for (int nt = 0; nt < 4; ++nt) { v8f z = {}; c[nt] = z; }
#pragma unroll
    for (int nt = 0; nt < 4; ++nt) {
#pragma unroll
      for (int jj = 0; jj < 2; ++jj) {
        v16h kb = ld_frag(Khd + (size_t)(t0 + nt * 16) * HEAD_DIM + jj * 32, HEAD_DIM);
        c[nt] = __builtin_amdgcn_wmma_f32_16x16x32_f16(false, qa[jj], false, kb,
                                                       (short)0, c[nt], false, false);
      }
    }
    // normalized probabilities -> LDS (transpose C-layout to A-layout) + head average
#pragma unroll
    for (int nt = 0; nt < 4; ++nt) {
#pragma unroll
      for (int i = 0; i < 8; ++i) {
        float p = __expf(c[nt][i] * inv_hd - mrow[i]) * lrow[i];
        lds_p[h][i + 8 * half][nt * 16 + col] = (_Float16)p;
        atomicAdd(&lds_ave[i + 8 * half][nt * 16 + col], p * 0.125f);
      }
    }
    __syncthreads();

    // P (16 x 64) @ V (64 x 64) accumulated into oacc
#pragma unroll
    for (int jj = 0; jj < 2; ++jj) {
      v16h pa = ld_frag(&lds_p[h][0][0] + jj * 32, 64);
#pragma unroll
      for (int nd = 0; nd < 4; ++nd) {
        v16h vb = ld_frag(Vhd + (size_t)(nd * 16) * SEQ + t0 + jj * 32, SEQ);
        oacc[nd] = __builtin_amdgcn_wmma_f32_16x16x32_f16(false, pa, false, vb,
                                                          (short)0, oacc[nd], false, false);
      }
    }

    // stream the averaged 16x64 tile to HBM once, then reset it
#pragma unroll
    for (int j = 0; j < 4; ++j) {
      int flat = threadIdx.x + j * 256;
      int r = flat >> 6, cc = flat & 63;
      ave[((size_t)b * SEQ + (s0 + r)) * SEQ + t0 + cc] = lds_ave[r][cc];
      lds_ave[r][cc] = 0.0f;
    }
    __syncthreads();
  }

  // per-head attention output (already normalized), PADDED concat-head layout
  // [b][s+1][D] f16 so the out-projection can use the unguarded path (taps=1
  // never reads pad rows).
#pragma unroll
  for (int nd = 0; nd < 4; ++nd) {
#pragma unroll
    for (int i = 0; i < 8; ++i) {
      const int s = s0 + i + 8 * half;
      Oh[((size_t)b * SEQP + (s + 1)) * D_MODEL + h * HEAD_DIM + nd * 16 + col] =
          (_Float16)oacc[nd][i];
    }
  }
}

// ---------------- launcher ----------------
extern "C" void kernel_launch(void* const* d_in, const int* in_sizes, int n_in,
                              void* d_out, int out_size, void* d_ws, size_t ws_size,
                              hipStream_t stream) {
  (void)in_sizes; (void)n_in; (void)out_size; (void)ws_size;
  const float* q  = (const float*)d_in[0];
  const float* k  = (const float*)d_in[1];
  const float* v  = (const float*)d_in[2];
  const float* Wq = (const float*)d_in[3];
  const float* bq = (const float*)d_in[4];
  const float* Wk = (const float*)d_in[5];
  const float* bk = (const float*)d_in[6];
  const float* Wv = (const float*)d_in[7];
  const float* bv = (const float*)d_in[8];
  const float* Wo = (const float*)d_in[9];
  const float* bo = (const float*)d_in[10];

  const size_t NX  = (size_t)BATCH * SEQ * D_MODEL;    // 4,194,304
  const size_t NXP = (size_t)BATCH * SEQP * D_MODEL;   // padded activations
  const size_t NW  = (size_t)D_MODEL * D_MODEL;        // 262,144

  char* ws = (char*)d_ws;
  size_t off = 0;
  auto alloc = [&](size_t bytes) -> char* {
    char* p = ws + off; off += (bytes + 255) & ~(size_t)255; return p;
  };
  _Float16* xq  = (_Float16*)alloc(NXP * 2);           // padded [b][s+2][D]
  _Float16* xk  = (_Float16*)alloc(NXP * 2);
  _Float16* xv  = (_Float16*)alloc(NXP * 2);
  _Float16* wqh = (_Float16*)alloc(NW * 3 * 2);
  _Float16* wkh = (_Float16*)alloc(NW * 3 * 2);
  _Float16* wvh = (_Float16*)alloc(NW * 2);
  _Float16* woh = (_Float16*)alloc(NW * 2);
  _Float16* Qh  = (_Float16*)alloc(NX * 2);            // [b,h,s,d]
  _Float16* Kh  = (_Float16*)alloc(NX * 2);            // [b,h,s,d]
  _Float16* Vt  = (_Float16*)alloc(NX * 2);            // [b,h,d,t]
  _Float16* Oh  = (_Float16*)alloc(NXP * 2);           // padded [b][s+2][D]

  float* out_p = (float*)d_out;
  float* ave_p = out_p + NX;                           // tuple order: out, ave_att

  const int CB = 256;
  // f32 -> padded f16 conversions (zero rows at each batch edge)
  cvt_pad_kernel<<<(int)((NXP + CB - 1) / CB), CB, 0, stream>>>(q, xq, (int)NXP);
  cvt_pad_kernel<<<(int)((NXP + CB - 1) / CB), CB, 0, stream>>>(k, xk, (int)NXP);
  cvt_pad_kernel<<<(int)((NXP + CB - 1) / CB), CB, 0, stream>>>(v, xv, (int)NXP);
  cvt_f16_kernel<<<(int)((NW + CB - 1) / CB), CB, 0, stream>>>(Wv, wvh, (int)NW);
  cvt_f16_kernel<<<(int)((NW + CB - 1) / CB), CB, 0, stream>>>(Wo, woh, (int)NW);
  repack_w3_kernel<<<(int)((NW + CB - 1) / CB), CB, 0, stream>>>(Wq, wqh);
  repack_w3_kernel<<<(int)((NW + CB - 1) / CB), CB, 0, stream>>>(Wk, wkh);

  // conv projections (3-tap for Q/K, 1-tap for V); 32x64 tile per wave
  dim3 pgrid(BATCH * SEQ / 32, D_MODEL / 64);
  proj_kernel<<<pgrid, 32, 0, stream>>>(xq, wqh, bq, Qh, 3, 0);
  proj_kernel<<<pgrid, 32, 0, stream>>>(xk, wkh, bk, Kh, 3, 0);
  proj_kernel<<<pgrid, 32, 0, stream>>>(xv, wvh, bv, Vt, 1, 1);

  // fused attention + averaged attention map
  attn_kernel<<<BATCH * SEQ / 16, 256, 0, stream>>>(Qh, Kh, Vt, Oh, ave_p);

  // output projection (f32 out)
  proj_kernel<<<pgrid, 32, 0, stream>>>(Oh, woh, bo, out_p, 1, 2);
}
